// MPNN_70265664962880
// MI455X (gfx1250) — compile-verified, matrix operand order checked by reading, not run on hardware
//
#include <hip/hip_runtime.h>
#include <math.h>

#define NATOM 50000
#define NEDGE 1600000
#define NWAVE 8
#define NORB  32
#define NHID  64
#define NTILE (NATOM / 16)   // 3125, exact
#define WFRAG 7168           // halves per MLP: W1 2048 + W2 4096 + W3 1024 (fragment order)

typedef __attribute__((ext_vector_type(16))) _Float16 v16h;
typedef __attribute__((ext_vector_type(8)))  float    v8f;

// ---------------------------------------------------------------- helpers
__device__ __forceinline__ float siluf(float x) { return x / (1.0f + __expf(-x)); }

// Pack a K x Nsrc f32 weight matrix into WMMA B-fragment order (f16),
// padded to Npad columns. Layout: ((kb*NB + nb)*32 + lane)*16 + h, so each
// lane's 16 halves are contiguous (one aligned 32B load per fragment).
__global__ void k_pack_frag(const float* __restrict__ src, _Float16* __restrict__ dst,
                            int K, int Nsrc, int Npad) {
    int i = blockIdx.x * blockDim.x + threadIdx.x;
    int NB = Npad / 16;
    int total = (K / 32) * NB * 32 * 16;
    if (i >= total) return;
    int h    = i & 15;
    int lane = (i >> 4) & 31;
    int blk  = i >> 9;            // kb*NB + nb
    int nb = blk % NB, kb = blk / NB;
    int b = (lane >= 16) ? 8 : 0;
    int k = kb * 32 + ((h < 8) ? (b + h) : (16 + b + (h - 8)));
    int n = nb * 16 + (lane & 15);
    dst[i] = (n < Nsrc) ? (_Float16)src[k * Nsrc + n] : (_Float16)0.0f;
}

// ---------------------------------------------------------------- embedding MLP: species -> center_coeff (NATOM x 24)
__global__ void k_emb(const float* __restrict__ species,
                      const float* __restrict__ W0, const float* __restrict__ b0,
                      const float* __restrict__ W1, const float* __restrict__ b1,
                      const float* __restrict__ W2, const float* __restrict__ b2,
                      float* __restrict__ cc) {
    int a = blockIdx.x * blockDim.x + threadIdx.x;
    if (a >= NATOM) return;
    float s = species[a];
    float h0[8], h1[8];
#pragma unroll
    for (int j = 0; j < 8; ++j) h0[j] = s * W0[j] + b0[j];
    {
        float m = 0.f; for (int j = 0; j < 8; ++j) m += h0[j]; m *= 0.125f;
        float v = 0.f; for (int j = 0; j < 8; ++j) { float d = h0[j] - m; v += d * d; } v *= 0.125f;
        float inv = rsqrtf(v + 1e-5f);
        for (int j = 0; j < 8; ++j) h0[j] = siluf((h0[j] - m) * inv);
    }
#pragma unroll
    for (int j = 0; j < 8; ++j) {
        float acc = b1[j];
        for (int k = 0; k < 8; ++k) acc += h0[k] * W1[k * 8 + j];
        h1[j] = acc;
    }
    {
        float m = 0.f; for (int j = 0; j < 8; ++j) m += h1[j]; m *= 0.125f;
        float v = 0.f; for (int j = 0; j < 8; ++j) { float d = h1[j] - m; v += d * d; } v *= 0.125f;
        float inv = rsqrtf(v + 1e-5f);
        for (int j = 0; j < 8; ++j) h1[j] = siluf((h1[j] - m) * inv);
    }
#pragma unroll
    for (int j = 0; j < 24; ++j) {
        float acc = b2[j];
        for (int k = 0; k < 8; ++k) acc += h1[k] * W2[k * 24 + j];
        cc[(size_t)a * 24 + j] = acc;
    }
}

// ---------------------------------------------------------------- edge precompute: cut, wd, scatter wc
__global__ void k_edge_pre(const int* __restrict__ nl,
                           const float* __restrict__ cart,
                           const float* __restrict__ shifts,
                           const float* __restrict__ nfac,
                           const float* __restrict__ cc,
                           float* __restrict__ cut,
                           float* __restrict__ wd,
                           float* __restrict__ wc) {
    int e = blockIdx.x * blockDim.x + threadIdx.x;
    if (e >= NEDGE) return;
    int i0 = nl[e], i1 = nl[NEDGE + e];
    float dv[3];
#pragma unroll
    for (int k = 0; k < 3; ++k)
        dv[k] = cart[i1 * 3 + k] - cart[i0 * 3 + k] + shifts[(size_t)e * 3 + k];
    float d = sqrtf(dv[0] * dv[0] + dv[1] * dv[1] + dv[2] * dv[2]);
    float t = 0.5f * __cosf(d * 0.78539816339744831f) + 0.5f; // pi/CUTOFF, CUTOFF=4
    float cutv = nfac[e] * t * t;
    cut[e] = cutv;
#pragma unroll
    for (int k = 0; k < 3; ++k) dv[k] *= cutv;
    // vectorized gathers of center_coeff rows (24 floats, 16B aligned)
    float c0v[24], c1v[24];
    const float4* p0 = (const float4*)(cc + (size_t)i0 * 24);
    const float4* p1 = (const float4*)(cc + (size_t)i1 * 24);
#pragma unroll
    for (int q = 0; q < 6; ++q) { ((float4*)c0v)[q] = p0[q]; ((float4*)c1v)[q] = p1[q]; }
    float wv[24];
#pragma unroll
    for (int j = 0; j < 8; ++j) {
        float coef = c0v[j] * c1v[j];
        float rs   = c0v[8 + j]  * c1v[8 + j];
        float al   = c0v[16 + j] * c1v[16 + j];
        float u = al * (d - rs);
        float w = coef * __expf(-u * u);
        wv[j]      = w * dv[0];
        wv[8 + j]  = w * dv[1];
        wv[16 + j] = w * dv[2];
    }
    float4* wde = (float4*)(wd + (size_t)e * 24);
#pragma unroll
    for (int q = 0; q < 6; ++q) wde[q] = ((float4*)wv)[q];
    float* wcb = wc + (size_t)i0 * 24;
#pragma unroll
    for (int q = 0; q < 24; ++q) atomicAdd(&wcb[q], wv[q]);
}

// ---------------------------------------------------------------- edge update (iteration): read wc_old, scatter wc_new
__global__ void k_edge_upd(const int* __restrict__ nl,
                           const float* __restrict__ itc,  // NATOM x 8
                           const float* __restrict__ cut,
                           const float* __restrict__ wd,
                           const float* __restrict__ wcold,
                           float* __restrict__ wcnew) {
    int e = blockIdx.x * blockDim.x + threadIdx.x;
    if (e >= NEDGE) return;
    int i0 = nl[e], i1 = nl[NEDGE + e];
    float cutv = cut[e];
    float icv[8], wov[24], wdv[24];
    const float4* ip = (const float4*)(itc + (size_t)i1 * 8);
    ((float4*)icv)[0] = ip[0]; ((float4*)icv)[1] = ip[1];
    const float4* op = (const float4*)(wcold + (size_t)i1 * 24);
    const float4* dp = (const float4*)(wd + (size_t)e * 24);
#pragma unroll
    for (int q = 0; q < 6; ++q) { ((float4*)wov)[q] = op[q]; ((float4*)wdv)[q] = dp[q]; }
    float* wcb = wcnew + (size_t)i0 * 24;
#pragma unroll
    for (int k = 0; k < 3; ++k)
#pragma unroll
        for (int j = 0; j < 8; ++j) {
            float v = icv[j] * wdv[k * 8 + j] + cutv * wov[k * 8 + j];
            atomicAdd(&wcb[k * 8 + j], v);
        }
}

// ---------------------------------------------------------------- density: per-atom spherical contraction
__global__ void k_dens(const float* __restrict__ wc,
                       const float* __restrict__ ccoef, // 3 x 8 x 32
                       float* __restrict__ density, int accumulate) {
    __shared__ float ccs[3 * NWAVE * NORB];
    for (int i = threadIdx.x; i < 3 * NWAVE * NORB; i += blockDim.x) ccs[i] = ccoef[i];
    __syncthreads();
    int a = blockIdx.x * blockDim.x + threadIdx.x;
    if (a >= NATOM) return;
    const int idxl[9] = {0, 1, 1, 1, 2, 2, 2, 2, 2};
    const float c0 = 0.28209479177387814f, c1 = 0.4886025119029199f;
    const float c2a = 1.0925484305920792f, c2b = 0.31539156525252005f, c2c = 0.5462742152960396f;
    float w[24];
    const float4* wp = (const float4*)(wc + (size_t)a * 24);
#pragma unroll
    for (int q = 0; q < 6; ++q) ((float4*)w)[q] = wp[q];
    float sph[9][8];
#pragma unroll
    for (int j = 0; j < 8; ++j) {
        float x = w[j], y = w[8 + j], z = w[16 + j];
        float r2 = x * x + y * y + z * z;
        sph[0][j] = c0;
        sph[1][j] = c1 * y;
        sph[2][j] = c1 * z;
        sph[3][j] = c1 * x;
        sph[4][j] = c2a * x * y;
        sph[5][j] = c2a * y * z;
        sph[6][j] = c2b * (3.0f * z * z - r2);
        sph[7][j] = c2a * x * z;
        sph[8][j] = c2c * (x * x - y * y);
    }
    for (int m = 0; m < NORB; ++m) {
        float s = 0.f;
#pragma unroll
        for (int k = 0; k < 9; ++k) {
            const float* cck = &ccs[(idxl[k] * NWAVE) * NORB + m];
            float t = 0.f;
#pragma unroll
            for (int j = 0; j < 8; ++j) t += sph[k][j] * cck[j * NORB];
            s += t * t;
        }
        size_t di = (size_t)a * NORB + m;
        density[di] = accumulate ? (density[di] + s) : s;
    }
}

// ---------------------------------------------------------------- layernorm+silu over one 64-wide row in LDS
__device__ __forceinline__ void ln_silu_row64(float* row) {
    float m = 0.f;
#pragma unroll
    for (int c = 0; c < NHID; ++c) m += row[c];
    m *= (1.0f / NHID);
    float v = 0.f;
#pragma unroll
    for (int c = 0; c < NHID; ++c) { float d = row[c] - m; v += d * d; }
    v *= (1.0f / NHID);
    float inv = rsqrtf(v + 1e-5f);
#pragma unroll
    for (int c = 0; c < NHID; ++c) row[c] = siluf((row[c] - m) * inv);
}

// A-fragment: 16 halves per lane from a f32 row: [base+0..7] and [base+16..23]
__device__ __forceinline__ v16h pack_a(const float* base) {
    float4 a0 = *(const float4*)(base);
    float4 a1 = *(const float4*)(base + 4);
    float4 a2 = *(const float4*)(base + 16);
    float4 a3 = *(const float4*)(base + 20);
    v16h a;
    a[0]  = (_Float16)a0.x; a[1]  = (_Float16)a0.y; a[2]  = (_Float16)a0.z; a[3]  = (_Float16)a0.w;
    a[4]  = (_Float16)a1.x; a[5]  = (_Float16)a1.y; a[6]  = (_Float16)a1.z; a[7]  = (_Float16)a1.w;
    a[8]  = (_Float16)a2.x; a[9]  = (_Float16)a2.y; a[10] = (_Float16)a2.z; a[11] = (_Float16)a2.w;
    a[12] = (_Float16)a3.x; a[13] = (_Float16)a3.y; a[14] = (_Float16)a3.z; a[15] = (_Float16)a3.w;
    return a;
}

// ---------------------------------------------------------------- WMMA MLP: X(Nx32) -> 64 -> 64 -> nout
// one wave per 16-row tile; f16 WMMA, f32 accumulate; weights async-staged to LDS
__global__ void __launch_bounds__(256)
k_mlp(const float* __restrict__ X,
      const _Float16* __restrict__ Wblk,  // WFRAG halves, fragment order
      const float* __restrict__ b1, const float* __restrict__ b2,
      const float* __restrict__ b3,
      int nout, float* __restrict__ Y, int ldy) {
    __shared__ __align__(32) float H[8][16 * NHID];   // 4KB per wave
    __shared__ __align__(32) _Float16 Wl[WFRAG];      // 14KB block-wide weights

    // ---- block-wide async weight staging: global -> LDS (ASYNCcnt path) ----
    {
        const unsigned long long gbase = (unsigned long long)(const void*)Wblk;
        for (int i = threadIdx.x; i < WFRAG / 8; i += 256) {   // 896 x b128
            unsigned int ldsa = (unsigned int)(size_t)(const void*)(Wl + i * 8);
            unsigned int goff = (unsigned int)(i * 16);
            asm volatile("global_load_async_to_lds_b128 %0, %1, %2"
                         :: "v"(ldsa), "v"(goff), "s"(gbase)
                         : "memory");
        }
#if __has_builtin(__builtin_amdgcn_s_wait_asynccnt)
        __builtin_amdgcn_s_wait_asynccnt(0);
#else
        asm volatile("s_wait_asynccnt 0" ::: "memory");
#endif
    }
    __syncthreads();

    const _Float16* W1f = Wl;          // kb=0, nb=0..3
    const _Float16* W2f = Wl + 2048;   // kb=0..1, nb=0..3
    const _Float16* W3f = Wl + 6144;   // kb=0..1, nb=0

    const int wave = threadIdx.x >> 5;
    const int lane = threadIdx.x & 31;
    const int tile = blockIdx.x * 8 + wave;
    const bool active = tile < NTILE;
    const int mloc = lane & 15;
    const int hi = (lane >> 4) & 1;
    const int kb8 = hi * 8;
    float* Ht = H[wave];

    v8f acc[4];

    // ---------------- GEMM1: X(16x32) x W1(32x64) ----------------
    if (active) {
        const float* Xt = X + (size_t)tile * 16 * NORB;
        v16h a = pack_a(Xt + mloc * NORB + kb8);
#pragma unroll
        for (int nb = 0; nb < 4; ++nb) {
            v16h bf = *(const v16h*)(W1f + (nb * 32 + lane) * 16);
            v8f c = {};
            c = __builtin_amdgcn_wmma_f32_16x16x32_f16(false, a, false, bf, (short)0, c,
                                                       false, false);
            acc[nb] = c;
        }
#pragma unroll
        for (int nb = 0; nb < 4; ++nb) {
            int n = nb * 16 + mloc;
            float bias = b1[n];
#pragma unroll
            for (int r = 0; r < 8; ++r) Ht[(r + hi * 8) * NHID + n] = acc[nb][r] + bias;
        }
    }
    __syncthreads();
    if (active && lane < 16) ln_silu_row64(Ht + lane * NHID);
    __syncthreads();

    // ---------------- GEMM2: H(16x64) x W2(64x64) ----------------
    if (active) {
        v16h a0 = pack_a(Ht + mloc * NHID + kb8);
        v16h a1 = pack_a(Ht + mloc * NHID + 32 + kb8);
#pragma unroll
        for (int nb = 0; nb < 4; ++nb) {
            v16h bf0 = *(const v16h*)(W2f + ((0 * 4 + nb) * 32 + lane) * 16);
            v16h bf1 = *(const v16h*)(W2f + ((1 * 4 + nb) * 32 + lane) * 16);
            v8f c = {};
            c = __builtin_amdgcn_wmma_f32_16x16x32_f16(false, a0, false, bf0, (short)0, c,
                                                       false, false);
            c = __builtin_amdgcn_wmma_f32_16x16x32_f16(false, a1, false, bf1, (short)0, c,
                                                       false, false);
            acc[nb] = c;
        }
#pragma unroll
        for (int nb = 0; nb < 4; ++nb) {
            int n = nb * 16 + mloc;
            float bias = b2[n];
#pragma unroll
            for (int r = 0; r < 8; ++r) Ht[(r + hi * 8) * NHID + n] = acc[nb][r] + bias;
        }
    }
    __syncthreads();
    if (active && lane < 16) ln_silu_row64(Ht + lane * NHID);
    __syncthreads();

    // ---------------- GEMM3: H(16x64) x W3(64x16 padded) ----------------
    if (active) {
        v16h a0 = pack_a(Ht + mloc * NHID + kb8);
        v16h a1 = pack_a(Ht + mloc * NHID + 32 + kb8);
        v16h bf0 = *(const v16h*)(W3f + (0 * 32 + lane) * 16);
        v16h bf1 = *(const v16h*)(W3f + (1 * 32 + lane) * 16);
        v8f c = {};
        c = __builtin_amdgcn_wmma_f32_16x16x32_f16(false, a0, false, bf0, (short)0, c,
                                                   false, false);
        c = __builtin_amdgcn_wmma_f32_16x16x32_f16(false, a1, false, bf1, (short)0, c,
                                                   false, false);
        if (mloc < nout) {
            float bias = b3[mloc];
#pragma unroll
            for (int r = 0; r < 8; ++r) {
                int row = tile * 16 + r + hi * 8;
                Y[(size_t)row * ldy + mloc] = c[r] + bias;
            }
        }
    }
}

// ---------------------------------------------------------------- final weighted reduction
__global__ void k_reduce(const float* __restrict__ ao, const float* __restrict__ cf,
                         float* __restrict__ out) {
    __shared__ float sm[256];
    float s = 0.f;
    for (int i = blockIdx.x * blockDim.x + threadIdx.x; i < NATOM; i += gridDim.x * blockDim.x)
        s += ao[i] * cf[i];
    sm[threadIdx.x] = s;
    __syncthreads();
    for (int off = 128; off > 0; off >>= 1) {
        if (threadIdx.x < off) sm[threadIdx.x] += sm[threadIdx.x + off];
        __syncthreads();
    }
    if (threadIdx.x == 0) atomicAdd(out, sm[0]);
}

// ================================================================ host
extern "C" void kernel_launch(void* const* d_in, const int* in_sizes, int n_in,
                              void* d_out, int out_size, void* d_ws, size_t ws_size,
                              hipStream_t stream) {
    (void)in_sizes; (void)n_in; (void)out_size;
    const float* cart    = (const float*)d_in[0];
    const int*   nl      = (const int*)d_in[1];
    const float* shifts  = (const float*)d_in[2];
    const float* cfac    = (const float*)d_in[3];
    const float* nfac    = (const float*)d_in[4];
    const float* species = (const float*)d_in[5];
    // params flattened as jax pytree (dict keys sorted): contracted_coeff, emb, iters, out
    const float* ccoef = (const float*)d_in[6];
    const float* eW0 = (const float*)d_in[7];  const float* eb0 = (const float*)d_in[8];
    const float* eW1 = (const float*)d_in[9];  const float* eb1 = (const float*)d_in[10];
    const float* eW2 = (const float*)d_in[11]; const float* eb2 = (const float*)d_in[12];
    // iters: indices 13 + 6*i + {0..5}; out: 31..36

    // ---- workspace layout ----
    char* ws = (char*)d_ws;
    size_t off = 0;
    auto alloc = [&](size_t bytes) -> char* {
        char* p = ws + off;
        off += (bytes + 255) & ~(size_t)255;
        return p;
    };
    float* centc  = (float*)alloc((size_t)NATOM * 24 * 4);
    float* cut    = (float*)alloc((size_t)NEDGE * 4);
    float* wd     = (float*)alloc((size_t)NEDGE * 24 * 4);
    float* wcA    = (float*)alloc((size_t)NATOM * 24 * 4);
    float* wcB    = (float*)alloc((size_t)NATOM * 24 * 4);
    float* dens   = (float*)alloc((size_t)NATOM * 32 * 4);
    float* itout  = (float*)alloc((size_t)NATOM * 8 * 4);
    float* aout   = (float*)alloc((size_t)NATOM * 4);
    _Float16* w16 = (_Float16*)alloc((size_t)4 * WFRAG * 2);
    if (off > ws_size) return;  // workspace too small (constant across calls)

    // ---- pack MLP weights into f16 WMMA fragment order ----
    for (int i = 0; i < 4; ++i) {
        int base = (i < 3) ? (13 + 6 * i) : 31;
        int nc3  = (i < 3) ? 8 : 1;
        _Float16* m = w16 + (size_t)i * WFRAG;
        k_pack_frag<<<(2048 + 255) / 256, 256, 0, stream>>>(
            (const float*)d_in[base + 0], m, 32, 64, 64);
        k_pack_frag<<<(4096 + 255) / 256, 256, 0, stream>>>(
            (const float*)d_in[base + 2], m + 2048, 64, 64, 64);
        k_pack_frag<<<(1024 + 255) / 256, 256, 0, stream>>>(
            (const float*)d_in[base + 4], m + 6144, 64, nc3, 16);
    }

    // ---- embedding ----
    k_emb<<<(NATOM + 255) / 256, 256, 0, stream>>>(species, eW0, eb0, eW1, eb1, eW2, eb2, centc);

    // ---- edge precompute + initial scatter ----
    hipMemsetAsync(wcA, 0, (size_t)NATOM * 24 * 4, stream);
    k_edge_pre<<<NEDGE / 256, 256, 0, stream>>>(nl, cart, shifts, nfac, centc, cut, wd, wcA);

    // ---- initial density ----
    k_dens<<<(NATOM + 255) / 256, 256, 0, stream>>>(wcA, ccoef, dens, 0);

    // ---- 3 message-passing iterations ----
    float* wc_cur = wcA;
    float* wc_nxt = wcB;
    const int mlp_blocks = (NTILE + 7) / 8;
    for (int it = 0; it < 3; ++it) {
        int base = 13 + 6 * it;
        _Float16* m = w16 + (size_t)it * WFRAG;
        k_mlp<<<mlp_blocks, 256, 0, stream>>>(dens, m,
                                              (const float*)d_in[base + 1],
                                              (const float*)d_in[base + 3],
                                              (const float*)d_in[base + 5],
                                              8, itout, 8);
        hipMemcpyAsync(wc_nxt, wc_cur, (size_t)NATOM * 24 * 4,
                       hipMemcpyDeviceToDevice, stream);
        k_edge_upd<<<NEDGE / 256, 256, 0, stream>>>(nl, itout, cut, wd, wc_cur, wc_nxt);
        float* tmp = wc_cur; wc_cur = wc_nxt; wc_nxt = tmp;
        k_dens<<<(NATOM + 255) / 256, 256, 0, stream>>>(wc_cur, ccoef, dens, 1);
    }

    // ---- output MLP + weighted reduction ----
    {
        _Float16* m = w16 + (size_t)3 * WFRAG;
        k_mlp<<<mlp_blocks, 256, 0, stream>>>(dens, m,
                                              (const float*)d_in[32],
                                              (const float*)d_in[34],
                                              (const float*)d_in[36],
                                              1, aout, 1);
    }
    hipMemsetAsync(d_out, 0, sizeof(float), stream);
    k_reduce<<<128, 256, 0, stream>>>(aout, cfac, (float*)d_out);
}
